// ObsEncoder_student_47261820125276
// MI455X (gfx1250) — matrix operand order repaired and checked by reading, not compiled, not run on hardware
//
#include <hip/hip_runtime.h>
#include <hip/hip_bf16.h>
#include <math.h>

// ---------------------------------------------------------------------------
// Fused ObsEncoder for MI455X (gfx1250, wave32, WMMA).
//
// One wave32 block owns 16 batch rows end-to-end. Dense GEMMs
// (self_emb@Ac/Lc/fcW, f1, f2) run on v_wmma_f32_16x16x32_f16 with f16
// operands / f32 accumulation. Tiny-K entity encoders + attention run on
// VALU with an online-softmax (flash) accumulation so entity embeddings are
// never materialized. All intermediates live in LDS / VGPRs: HBM traffic is
// exactly inputs-once + outputs-once.
//
// Round-3 change vs round-2:
//  * __launch_bounds__(32, 4) on the main kernel: round-2 disasm still
//    allocated >256 VGPRs (s_set_vgpr_msb in the hot loop) because the
//    compiler prefetched B-fragments arbitrarily deep. Capping at ~256 VGPRs
//    (1024/4 per SIMD) buys ~4 co-resident waves per SIMD so other blocks'
//    WMMA phases overlap this block's VALU-bound attention phase, and kills
//    the per-loop s_set_vgpr_msb SALU overhead. Attention needs ~190 live
//    VGPRs, so the cap trims only GEMM prefetch depth, not spills.
//
// Workspace: pre-swizzled f16 WMMA B-fragments for the 5 weight matrices.
//   224 fragments x 32 lanes x 16 halves x 2B = 229,376 bytes needed.
// ---------------------------------------------------------------------------

typedef __attribute__((ext_vector_type(16))) _Float16 v16h;
typedef __attribute__((ext_vector_type(8)))  _Float16 v8h;
typedef __attribute__((ext_vector_type(8)))  float    v8f;

#define NT     7    // 112 / 16 output-column tiles (H=100 padded to 112)
#define KT_H   4    // 128 / 32 K-steps for K=100 (padded to 128)
#define KT_X   16   // 512 / 32 K-steps for K=400 (4 segments of 128)

#define FR_AC  0
#define FR_LC  28
#define FR_FC  56
#define FR_F2  84
#define FR_F1  112
#define FR_TOTAL 224   // total fragments in workspace

struct Params {
    const float *in;
    const float *sW, *sb, *sg, *sbe;
    const float *oW, *ob, *og, *obe;
    const float *lW, *lb, *lg, *lbe;
    const float *gW, *gb, *gg, *gbe;
    const float *fcb, *fcg, *fcbe;
    const float *f1b, *f1g, *f1be;
    const float *f2b, *f2g, *f2be;
    const _Float16 *frags;
    float *out;
};

// fast tanh: sign(x) * (1-e)/(1+e), e = exp(-2|x|).
// v_exp_f32 + v_rcp_f32; accuracy ~1e-6 rel, far above downstream f16 quant.
__device__ __forceinline__ float tanh_fast(float x) {
    float ax = __builtin_fabsf(x);
    float e  = __expf(-2.f * ax);
    float t  = (1.f - e) * __builtin_amdgcn_rcpf(1.f + e);
    return __builtin_copysignf(t, x);
}

// ---------------------------------------------------------------------------
// Prep: convert + swizzle weights into WMMA B-fragment order.
// B-fragment order chosen here (16-bit B, 32x16; contiguous k, same as the A side): lane l holds column n = l%16;
// lanes 0-15 carry K 0..15, lanes 16-31 carry K 16..31; element e -> k_local =
// (l/16)*16 + e. One fragment = 32 lanes x 16 halves, stored contiguously.
// ---------------------------------------------------------------------------
__global__ __launch_bounds__(32)
void obsenc_prep(const float* Ac, const float* Lc, const float* fcW,
                 const float* f2W, const float* f1W, _Float16* ws16) {
    int f    = blockIdx.x;          // fragment id, 0..223
    int lane = threadIdx.x;         // 0..31
    const float* src;
    int loc; bool isF1 = false;
    if      (f < FR_LC) { src = Ac;  loc = f - FR_AC; }
    else if (f < FR_FC) { src = Lc;  loc = f - FR_LC; }
    else if (f < FR_F2) { src = fcW; loc = f - FR_FC; }
    else if (f < FR_F1) { src = f2W; loc = f - FR_F2; }
    else                { src = f1W; loc = f - FR_F1; isF1 = true; }
    int kt = loc / NT;
    int nt = loc % NT;
    int nl = lane & 15, kh = lane >> 4;
    int n  = nt * 16 + nl;
    v16h out;
#pragma unroll
    for (int e = 0; e < 16; ++e) {
        int k = kt * 32 + kh * 16 + e;
        float v = 0.f;
        if (n < 100) {
            if (!isF1) {
                if (k < 100) v = src[k * 100 + n];
            } else {
                int seg = k >> 7, j = k & 127;     // 4 segments of 128 <- 100
                if (j < 100) v = src[(seg * 100 + j) * 100 + n];
            }
        }
        out[e] = (_Float16)v;
    }
    *(v16h*)(ws16 + ((size_t)f * 32 + lane) * 16) = out;
}

// ---------------------------------------------------------------------------
// Device helpers
// ---------------------------------------------------------------------------
__device__ __forceinline__ v16h load_bfrag(const _Float16* frags, int f, int lane) {
    return *(const v16h*)(frags + ((size_t)f * 32 + lane) * 16);
}

// A-fragment from row-major LDS [m][k] (16-bit A 16x32 layout, contiguous k):
// lane l (m=l%16, half=l/16) holds k = half*16 + 0..15, matching the B
// fragments -> two contiguous ds_load_b128.
__device__ __forceinline__ v16h gather_afrag(const _Float16* lds, int strideH,
                                             int kt, int lane) {
    int m = lane & 15, half = lane >> 4;
    const _Float16* base = lds + m * strideH + kt * 32 + half * 16;
    v8h lo = *(const v8h*)(base);         // k_local = half*16 + 0..7
    v8h hi = *(const v8h*)(base + 8);     // k_local = half*16 + 8..15
    v16h r;
#pragma unroll
    for (int e = 0; e < 8; ++e) { r[e] = lo[e]; r[e + 8] = hi[e]; }
    return r;
}

// C/D 16x16 f32 layout: lane l holds col n=l%16; VGPR i -> row m = (l/16)*8+i.
__device__ __forceinline__ void store_cfrag(float* dst, int stride, int nt,
                                            v8f c, int lane) {
    int nl = lane & 15, half = lane >> 4;
#pragma unroll
    for (int i = 0; i < 8; ++i)
        dst[(half * 8 + i) * stride + nt * 16 + nl] = c[i];
}

// tanh + LayerNorm over the 100 valid columns of tmp[16][112].
// 2 lanes per row (sub = lane/16), 50 elements each; stats via shfl_xor(16).
__device__ __forceinline__ void tanh_ln_store(const float* tmp, const float* bias,
                                              const float* g, const float* be,
                                              _Float16* dst16, int stride16, int segOff,
                                              float* dstG, int lane) {
    int row = lane & 15, sub = lane >> 4;
    float v[50]; float s = 0.f, s2 = 0.f;
#pragma unroll
    for (int i = 0; i < 50; ++i) {
        int h = sub * 50 + i;
        float t = tanh_fast(tmp[row * 112 + h] + bias[h]);
        v[i] = t; s += t; s2 += t * t;
    }
    s  += __shfl_xor(s, 16, 32);
    s2 += __shfl_xor(s2, 16, 32);
    float mean = s * 0.01f;
    float inv  = rsqrtf(s2 * 0.01f - mean * mean + 1e-5f);
#pragma unroll
    for (int i = 0; i < 50; ++i) {
        int h = sub * 50 + i;
        float o = (v[i] - mean) * inv * g[h] + be[h];
        if (dst16) dst16[row * stride16 + segOff + h] = (_Float16)o;
        else       dstG[row * 100 + h] = o;
    }
}

// One attention branch, streamed with online softmax.
// For each entity: tiny-K linear -> tanh -> LN -> dot with q -> flash update.
template <int K, int NE>
__device__ __forceinline__ void attn_branch(const float* inbuf, int featBase,
                                            const float* W, const float* b,
                                            const float* g, const float* be,
                                            const float* qbuf,
                                            _Float16* xcat, int segOff, int lane) {
    int row = lane & 15, sub = lane >> 4;
    float q[50], acc[50];
#pragma unroll
    for (int i = 0; i < 50; ++i) {
        q[i]   = qbuf[row * 112 + sub * 50 + i];
        acc[i] = 0.f;
    }
    float mrun = -1e30f, denom = 0.f;
    for (int n = 0; n < NE; ++n) {
        float f[K];
#pragma unroll
        for (int k = 0; k < K; ++k)
            f[k] = inbuf[row * 114 + featBase + n * K + k];
        float t[50]; float s = 0.f, s2 = 0.f;
#pragma unroll
        for (int i = 0; i < 50; ++i) {
            int h = sub * 50 + i;
            float lin = b[h];
#pragma unroll
            for (int k = 0; k < K; ++k) lin += f[k] * W[k * 100 + h];
            float tt = tanh_fast(lin);
            t[i] = tt; s += tt; s2 += tt * tt;
        }
        s  += __shfl_xor(s, 16, 32);
        s2 += __shfl_xor(s2, 16, 32);
        float mean = s * 0.01f;
        float inv  = rsqrtf(s2 * 0.01f - mean * mean + 1e-5f);
        float dot = 0.f;
#pragma unroll
        for (int i = 0; i < 50; ++i) {
            int h = sub * 50 + i;
            float e = (t[i] - mean) * inv * g[h] + be[h];
            t[i] = e;
            dot += q[i] * e;
        }
        dot += __shfl_xor(dot, 16, 32);
        float mnew = fmaxf(mrun, dot);
        float c = __expf(mrun - mnew);
        float w = __expf(dot - mnew);
        denom = denom * c + w;
        mrun  = mnew;
#pragma unroll
        for (int i = 0; i < 50; ++i) acc[i] = acc[i] * c + w * t[i];
    }
    float invd = __builtin_amdgcn_rcpf(denom);
#pragma unroll
    for (int i = 0; i < 50; ++i)
        xcat[row * 512 + segOff + sub * 50 + i] = (_Float16)(acc[i] * invd);
}

// ---------------------------------------------------------------------------
// Main fused kernel: 1 wave32 block per 16-row batch tile.
// __launch_bounds__(32, 4): cap ~256 VGPRs -> ~4 waves/SIMD co-residency.
// ---------------------------------------------------------------------------
__global__ __launch_bounds__(32, 4)
void obsenc_main(Params p) {
    __shared__ float    inbuf[16 * 114];   // raw observations
    __shared__ _Float16 sfh  [16 * 128];   // self_emb f16, K-padded (WMMA A src)
    __shared__ float    qa_s [16 * 112];   // self_emb @ Ac
    __shared__ float    ql_s [16 * 112];   // self_emb @ Lc
    __shared__ float    tmp_s[16 * 112];   // WMMA C dump / LN input
    __shared__ _Float16 xcat [16 * 512];   // [gi|va|vl|vg], each seg 128 (100 valid)
    __shared__ _Float16 hbuf [16 * 128];   // f1 output f16, K-padded

    int lane = threadIdx.x;
    int r0   = blockIdx.x * 16;

    // zero f16 pad regions once
    for (int idx = lane; idx < 16 * 128; idx += 32) { sfh[idx] = (_Float16)0.f; hbuf[idx] = (_Float16)0.f; }
    for (int idx = lane; idx < 16 * 512; idx += 32) xcat[idx] = (_Float16)0.f;
    // stage 16 rows of input (contiguous, coalesced)
    for (int idx = lane; idx < 16 * 114; idx += 32)
        inbuf[idx] = p.in[(size_t)r0 * 114 + idx];
    __syncthreads();

    // ---- self encoder: LN(tanh(x[:, :4] @ sW + sb)) -> sfh (f16) ----
    {
        int row = lane & 15, sub = lane >> 4;
        float x0 = inbuf[row * 114 + 0], x1 = inbuf[row * 114 + 1];
        float x2 = inbuf[row * 114 + 2], x3 = inbuf[row * 114 + 3];
        float v[50]; float s = 0.f, s2 = 0.f;
#pragma unroll
        for (int i = 0; i < 50; ++i) {
            int h = sub * 50 + i;
            float lin = p.sb[h] + x0 * p.sW[h] + x1 * p.sW[100 + h]
                                + x2 * p.sW[200 + h] + x3 * p.sW[300 + h];
            float t = tanh_fast(lin);
            v[i] = t; s += t; s2 += t * t;
        }
        s  += __shfl_xor(s, 16, 32);
        s2 += __shfl_xor(s2, 16, 32);
        float mean = s * 0.01f;
        float inv  = rsqrtf(s2 * 0.01f - mean * mean + 1e-5f);
#pragma unroll
        for (int i = 0; i < 50; ++i) {
            int h = sub * 50 + i;
            sfh[row * 128 + h] = (_Float16)((v[i] - mean) * inv * p.sg[h] + p.sbe[h]);
        }
    }
    __syncthreads();

    // ---- qa / ql / fc : three GEMMs sharing the self_emb A fragments ----
    {
        v16h aS[KT_H];
#pragma unroll
        for (int kt = 0; kt < KT_H; ++kt) aS[kt] = gather_afrag(sfh, 128, kt, lane);
        for (int nt = 0; nt < NT; ++nt) {
            v8f cq = {}, cl = {}, cf = {};
#pragma unroll
            for (int kt = 0; kt < KT_H; ++kt) {
                v16h bq = load_bfrag(p.frags, FR_AC + kt * NT + nt, lane);
                v16h bl = load_bfrag(p.frags, FR_LC + kt * NT + nt, lane);
                v16h bf = load_bfrag(p.frags, FR_FC + kt * NT + nt, lane);
                cq = __builtin_amdgcn_wmma_f32_16x16x32_f16(false, aS[kt], false, bq, (short)0, cq, false, false);
                cl = __builtin_amdgcn_wmma_f32_16x16x32_f16(false, aS[kt], false, bl, (short)0, cl, false, false);
                cf = __builtin_amdgcn_wmma_f32_16x16x32_f16(false, aS[kt], false, bf, (short)0, cf, false, false);
            }
            store_cfrag(qa_s, 112, nt, cq, lane);
            store_cfrag(ql_s, 112, nt, cl, lane);
            store_cfrag(tmp_s, 112, nt, cf, lane);
        }
    }
    __syncthreads();

    // ---- gi = LN(tanh(self_emb @ fcW + fcb)) -> xcat seg 0 ----
    tanh_ln_store(tmp_s, p.fcb, p.fcg, p.fcbe, xcat, 512, 0, nullptr, lane);
    __syncthreads();

    // ---- attention branches (note: ql reused for goals — bug preserved) ----
    attn_branch<2, 15>(inbuf, 52, p.oW, p.ob, p.og, p.obe, qa_s, xcat, 128, lane); // others
    attn_branch<3, 16>(inbuf,  4, p.lW, p.lb, p.lg, p.lbe, ql_s, xcat, 256, lane); // landmarks
    attn_branch<2, 16>(inbuf, 82, p.gW, p.gb, p.gg, p.gbe, ql_s, xcat, 384, lane); // goals
    __syncthreads();

    // ---- f1 : [16,512] @ [512,112], kt-outer to keep VGPR use bounded ----
    {
        v8f acc[NT];
        v8f zed = {};
#pragma unroll
        for (int nt = 0; nt < NT; ++nt) acc[nt] = zed;
        for (int kt = 0; kt < KT_X; ++kt) {
            v16h a = gather_afrag(xcat, 512, kt, lane);
#pragma unroll
            for (int nt = 0; nt < NT; ++nt) {
                v16h b = load_bfrag(p.frags, FR_F1 + kt * NT + nt, lane);
                acc[nt] = __builtin_amdgcn_wmma_f32_16x16x32_f16(false, a, false, b, (short)0, acc[nt], false, false);
            }
        }
#pragma unroll
        for (int nt = 0; nt < NT; ++nt) store_cfrag(tmp_s, 112, nt, acc[nt], lane);
    }
    __syncthreads();
    tanh_ln_store(tmp_s, p.f1b, p.f1g, p.f1be, hbuf, 128, 0, nullptr, lane);
    __syncthreads();

    // ---- f2 : [16,128] @ [128,112] ----
    {
        v16h aH[KT_H];
#pragma unroll
        for (int kt = 0; kt < KT_H; ++kt) aH[kt] = gather_afrag(hbuf, 128, kt, lane);
        for (int nt = 0; nt < NT; ++nt) {
            v8f c = {};
#pragma unroll
            for (int kt = 0; kt < KT_H; ++kt) {
                v16h b = load_bfrag(p.frags, FR_F2 + kt * NT + nt, lane);
                c = __builtin_amdgcn_wmma_f32_16x16x32_f16(false, aH[kt], false, b, (short)0, c, false, false);
            }
            store_cfrag(tmp_s, 112, nt, c, lane);
        }
    }
    __syncthreads();
    // final enc -> f32 global output
    tanh_ln_store(tmp_s, p.f2b, p.f2g, p.f2be, nullptr, 0, 0,
                  p.out + (size_t)r0 * 100, lane);
}

// ---------------------------------------------------------------------------
extern "C" void kernel_launch(void* const* d_in, const int* in_sizes, int n_in,
                              void* d_out, int out_size, void* d_ws, size_t ws_size,
                              hipStream_t stream) {
    // setup_inputs() dict order
    const float* inputs = (const float*)d_in[0];
    // d_in[1] = agent_num (int, ==16; shapes below assume it)
    Params p;
    p.in  = inputs;
    p.sW  = (const float*)d_in[2];  p.sb  = (const float*)d_in[3];
    p.sg  = (const float*)d_in[4];  p.sbe = (const float*)d_in[5];
    p.oW  = (const float*)d_in[6];  p.ob  = (const float*)d_in[7];
    p.og  = (const float*)d_in[8];  p.obe = (const float*)d_in[9];
    p.lW  = (const float*)d_in[10]; p.lb  = (const float*)d_in[11];
    p.lg  = (const float*)d_in[12]; p.lbe = (const float*)d_in[13];
    p.gW  = (const float*)d_in[14]; p.gb  = (const float*)d_in[15];
    p.gg  = (const float*)d_in[16]; p.gbe = (const float*)d_in[17];
    const float* Ac  = (const float*)d_in[18];
    const float* Lc  = (const float*)d_in[19];
    const float* fcW = (const float*)d_in[20];
    p.fcb = (const float*)d_in[21]; p.fcg = (const float*)d_in[22]; p.fcbe = (const float*)d_in[23];
    const float* f1W = (const float*)d_in[24];
    p.f1b = (const float*)d_in[25]; p.f1g = (const float*)d_in[26]; p.f1be = (const float*)d_in[27];
    const float* f2W = (const float*)d_in[28];
    p.f2b = (const float*)d_in[29]; p.f2g = (const float*)d_in[30]; p.f2be = (const float*)d_in[31];

    _Float16* ws16 = (_Float16*)d_ws;   // needs >= 229,376 bytes
    p.frags = ws16;
    p.out   = (float*)d_out;

    // 1) swizzle weights into WMMA B-fragments (tiny; L2-resident afterwards)
    obsenc_prep<<<FR_TOTAL, 32, 0, stream>>>(Ac, Lc, fcW, f2W, f1W, ws16);

    // 2) fused forward pass: one wave32 per 16 batch rows
    int B = in_sizes[0] / 114;
    obsenc_main<<<B / 16, 32, 0, stream>>>(p);
}
